// ParticleAttentionBlock_57397942944342
// MI455X (gfx1250) — matrix operand order repaired and checked by reading, not compiled, not run on hardware
//
#include <hip/hip_runtime.h>
#include <hip/hip_bf16.h>

#define B_SZ   8
#define NSEQ   1024
#define DMODEL 512
#define NHEADS 8
#define HDIM   32
#define DFF    2048
#define NROWS  (B_SZ * NSEQ)      // 8192
#define LAM0   0.2f               // LAMBDA_INIT for depth 1
#define EPS_LN 1e-5f
#define NEGINF (-1e30f)

typedef __attribute__((ext_vector_type(8)))  _Float16 v8h;
typedef __attribute__((ext_vector_type(16))) _Float16 v16h;
typedef __attribute__((ext_vector_type(8)))  float    v8f;

// ---------------- CDNA5 async global->LDS copy (ASYNCcnt-tracked) -------------
// Copies 16B from a per-lane global address directly into LDS, no VGPR transit.
__device__ __forceinline__ void async_b128(unsigned lds_off, const void* gptr) {
    asm volatile("global_load_async_to_lds_b128 %0, %1, off"
                 :: "v"(lds_off), "v"(gptr) : "memory");
}
__device__ __forceinline__ void wait_async() {
    asm volatile("s_wait_asynccnt 0x0" ::: "memory");
}
__device__ __forceinline__ unsigned lds_addr(const void* p) {
    // generic LDS address: addr[31:0] is the LDS byte offset (ISA aperture rule)
    return (unsigned)(uintptr_t)p;
}

// ---------------- WMMA fragment helpers (16x16x32 f16, wave32) ----------------
// A (16x32 MxK): lane L -> row m=L&15, half-sel h=L>>4.
//   halfs[0..15] = A[m][h*16 + 0..15]
__device__ __forceinline__ v16h afrag_from(const _Float16* row, int h) {
    v8h lo = *(const v8h*)(row + h * 16);
    v8h hi = *(const v8h*)(row + h * 16 + 8);
    v16h r;
#pragma unroll
    for (int i = 0; i < 8; ++i) { r[i] = lo[i]; r[i + 8] = hi[i]; }
    return r;
}
// B (32x16 KxN): lane L -> col n=L&15, h=L>>4; halfs[j] = B[h*16 + j][n].
__device__ __forceinline__ v16h bfrag_from(const _Float16* p) {
    v8h lo = *(const v8h*)(p);
    v8h hi = *(const v8h*)(p + 8);
    v16h r;
#pragma unroll
    for (int i = 0; i < 8; ++i) { r[i] = lo[i]; r[i + 8] = hi[i]; }
    return r;
}
__device__ __forceinline__ v8f wmma_f16(v16h a, v16h b, v8f c) {
    return __builtin_amdgcn_wmma_f32_16x16x32_f16(false, a, false, b, (short)0, c,
                                                  false, false);
}

// ---------------- generic NT GEMM: C[M,N] = A[M,K] * B[N,K]^T -----------------
// block tile 128(M) x 128(N), 8 waves, each wave 32x64 (2x4 WMMA tiles).
// Double-buffered LDS staged via async global->LDS copies.
template <bool HALF_OUT, bool HAS_BIAS, bool HAS_RES>
__global__ void __launch_bounds__(256)
gemm_nt_kernel(const _Float16* __restrict__ A, const _Float16* __restrict__ Bm,
               int M, int N, int K,
               float* __restrict__ Cf, _Float16* __restrict__ Ch,
               const float* __restrict__ bias, const float* __restrict__ resid)
{
    __shared__ _Float16 As[2][128][32];
    __shared__ _Float16 Bs[2][128][32];
    const int tid  = threadIdx.x;
    const int m0   = blockIdx.y * 128;
    const int n0   = blockIdx.x * 128;
    const int wid  = tid >> 5;
    const int lane = tid & 31;
    const int wm   = (wid & 3) * 32;
    const int wn   = (wid >> 2) * 64;
    const int nl   = lane & 15;
    const int hh   = lane >> 4;

    const int sr = tid >> 1;          // staging row 0..127
    const int sc = (tid & 1) * 16;    // staging col (halfs)

    auto stage = [&](int buf, int k0) {
        const _Float16* asrc = A + (size_t)(m0 + sr) * K + k0 + sc;
        unsigned al = lds_addr(&As[buf][sr][sc]);
        async_b128(al,      asrc);
        async_b128(al + 16, asrc + 8);
        const _Float16* bsrc = Bm + (size_t)(n0 + sr) * K + k0 + sc;
        unsigned bl = lds_addr(&Bs[buf][sr][sc]);
        async_b128(bl,      bsrc);
        async_b128(bl + 16, bsrc + 8);
    };

    v8f acc[2][4];
#pragma unroll
    for (int i = 0; i < 2; ++i)
#pragma unroll
        for (int j = 0; j < 4; ++j) acc[i][j] = {};

    stage(0, 0);
    wait_async();
    __syncthreads();

    const int nk = K >> 5;
    for (int kt = 0; kt < nk; ++kt) {
        const int cur = kt & 1;
        if (kt + 1 < nk) stage(cur ^ 1, (kt + 1) << 5);   // async prefetch next tile

        v16h af[2], bf[4];
#pragma unroll
        for (int i = 0; i < 2; ++i) af[i] = afrag_from(&As[cur][wm + i * 16 + nl][0], hh);
#pragma unroll
        for (int j = 0; j < 4; ++j) bf[j] = bfrag_from(&Bs[cur][wn + j * 16 + nl][hh * 16]);
#pragma unroll
        for (int i = 0; i < 2; ++i)
#pragma unroll
            for (int j = 0; j < 4; ++j) acc[i][j] = wmma_f16(af[i], bf[j], acc[i][j]);

        wait_async();        // next buffer resident in LDS
        __syncthreads();     // all waves done reading current buffer
    }

    // epilogue: reg r holds row (wm + i*16 + hh*8 + r), col (wn + j*16 + nl)
#pragma unroll
    for (int i = 0; i < 2; ++i)
#pragma unroll
        for (int j = 0; j < 4; ++j)
#pragma unroll
            for (int r = 0; r < 8; ++r) {
                int row = m0 + wm + i * 16 + hh * 8 + r;
                int col = n0 + wn + j * 16 + nl;
                float v = acc[i][j][r];
                if (HAS_BIAS) v += bias[col];
                if (HAS_RES)  v += resid[(size_t)row * N + col];
                if (HALF_OUT) Ch[(size_t)row * N + col] = (_Float16)v;
                else          Cf[(size_t)row * N + col] = v;
            }
}

// ------------- fused differential attention (flash-style, WMMA) ---------------
// grid: B*NHEADS*(NSEQ/128) blocks, 256 threads (8 waves); wave = 16 query rows
__global__ void __launch_bounds__(256)
diff_attn_kernel(const _Float16* __restrict__ Q, const _Float16* __restrict__ Kx,
                 const _Float16* __restrict__ V,
                 const float* __restrict__ lq1, const float* __restrict__ lk1,
                 const float* __restrict__ lq2, const float* __restrict__ lk2,
                 const float* __restrict__ subw, _Float16* __restrict__ out)
{
    __shared__ _Float16 K1s[64][32];
    __shared__ _Float16 K2s[64][32];
    __shared__ _Float16 VTs[64][64];        // [feature d][key r]
    __shared__ _Float16 Ps[8][2][16][64];   // per-wave P tiles, both maps

    const int tid  = threadIdx.x;
    const int wid  = tid >> 5;
    const int lane = tid & 31;
    const int nl   = lane & 15;
    const int hh   = lane >> 4;

    const int bx   = blockIdx.x;
    const int mb   = bx & 7;
    const int head = (bx >> 3) & 7;
    const int b    = bx >> 6;
    const int row0 = b * NSEQ + mb * 128 + wid * 16;
    const int qh1  = head * 2, qh2 = head * 2 + 1;

    // lambda = exp(lq1.lk1) - exp(lq2.lk2) + LAM0  (cheap, redundant per lane)
    float d1 = 0.f, d2 = 0.f;
#pragma unroll
    for (int i = 0; i < HDIM; ++i) { d1 += lq1[i] * lk1[i]; d2 += lq2[i] * lk2[i]; }
    const float lam = __expf(d1) - __expf(d2) + LAM0;

    // Q fragments for this wave's 16 rows (both sub-heads)
    v16h q1f = afrag_from(Q + (size_t)(row0 + nl) * DMODEL + qh1 * HDIM, hh);
    v16h q2f = afrag_from(Q + (size_t)(row0 + nl) * DMODEL + qh2 * HDIM, hh);

    v8f o1[4], o2[4];
#pragma unroll
    for (int c = 0; c < 4; ++c) { o1[c] = {}; o2[c] = {}; }
    float m1[8], m2[8], l1[8], l2[8];
#pragma unroll
    for (int r = 0; r < 8; ++r) { m1[r] = NEGINF; m2[r] = NEGINF; l1[r] = 0.f; l2[r] = 0.f; }

    const float scale = 0.17677669529663687f;   // 1/sqrt(32)

    for (int ct = 0; ct < NSEQ / 64; ++ct) {
        const int cbase = ct * 64;
        __syncthreads();
        {   // async-stage K1/K2 (64x32 each); manual transpose-stage V
            int r  = tid >> 2;
            int cb = (tid & 3) * 8;
            const _Float16* k1src = Kx + (size_t)(b * NSEQ + cbase + r) * DMODEL + qh1 * HDIM + cb;
            const _Float16* k2src = Kx + (size_t)(b * NSEQ + cbase + r) * DMODEL + qh2 * HDIM + cb;
            async_b128(lds_addr(&K1s[r][cb]), k1src);
            async_b128(lds_addr(&K2s[r][cb]), k2src);
            int dbase = (tid & 3) * 16;
            const _Float16* vsrc = V + (size_t)(b * NSEQ + cbase + r) * DMODEL + head * 64 + dbase;
            v8h va = *(const v8h*)(vsrc);
            v8h vb = *(const v8h*)(vsrc + 8);
#pragma unroll
            for (int j = 0; j < 8; ++j) { VTs[dbase + j][r] = va[j]; VTs[dbase + 8 + j][r] = vb[j]; }
        }
        wait_async();
        __syncthreads();

        // scores for both maps: 16 rows x 64 keys, K-dim = 32 (single WMMA each)
        v8f s1[4], s2[4];
#pragma unroll
        for (int t = 0; t < 4; ++t) {
            v16h k1f = bfrag_from(&K1s[t * 16 + nl][hh * 16]);
            v16h k2f = bfrag_from(&K2s[t * 16 + nl][hh * 16]);
            v8f z = {};
            s1[t] = wmma_f16(q1f, k1f, z);
            s2[t] = wmma_f16(q2f, k2f, z);
        }

        // dual online softmax; row (r + 8*hh) lives in reg r across the 16-lane half
#pragma unroll
        for (int r = 0; r < 8; ++r) {
            float tm1 = NEGINF, tm2 = NEGINF;
#pragma unroll
            for (int t = 0; t < 4; ++t) {
                s1[t][r] *= scale; s2[t][r] *= scale;
                tm1 = fmaxf(tm1, s1[t][r]); tm2 = fmaxf(tm2, s2[t][r]);
            }
#pragma unroll
            for (int msk = 8; msk >= 1; msk >>= 1) {
                tm1 = fmaxf(tm1, __shfl_xor(tm1, msk, 32));
                tm2 = fmaxf(tm2, __shfl_xor(tm2, msk, 32));
            }
            float nm1 = fmaxf(m1[r], tm1), nm2 = fmaxf(m2[r], tm2);
            float f1 = __expf(m1[r] - nm1), f2 = __expf(m2[r] - nm2);
            m1[r] = nm1; m2[r] = nm2;
            l1[r] *= f1; l2[r] *= f2;
#pragma unroll
            for (int c = 0; c < 4; ++c) { o1[c][r] *= f1; o2[c][r] *= f2; }
            float rs1 = 0.f, rs2 = 0.f;
#pragma unroll
            for (int t = 0; t < 4; ++t) {
                float p1 = __expf(s1[t][r] - nm1);
                float p2 = __expf(s2[t][r] - nm2);
                rs1 += p1; rs2 += p2;
                Ps[wid][0][hh * 8 + r][t * 16 + nl] = (_Float16)p1;
                Ps[wid][1][hh * 8 + r][t * 16 + nl] = (_Float16)p2;
            }
#pragma unroll
            for (int msk = 8; msk >= 1; msk >>= 1) {
                rs1 += __shfl_xor(rs1, msk, 32);
                rs2 += __shfl_xor(rs2, msk, 32);
            }
            l1[r] += rs1; l2[r] += rs2;
        }
        __syncthreads();

        // O += P * V  (K-dim 64 = two 32-steps; 64 output features = 4 tiles)
        v16h p1f[2], p2f[2];
#pragma unroll
        for (int kk = 0; kk < 2; ++kk) {
            p1f[kk] = afrag_from(&Ps[wid][0][nl][kk * 32], hh);
            p2f[kk] = afrag_from(&Ps[wid][1][nl][kk * 32], hh);
        }
#pragma unroll
        for (int c = 0; c < 4; ++c)
#pragma unroll
            for (int kk = 0; kk < 2; ++kk) {
                v16h vf = bfrag_from(&VTs[c * 16 + nl][kk * 32 + hh * 16]);
                o1[c] = wmma_f16(p1f[kk], vf, o1[c]);
                o2[c] = wmma_f16(p2f[kk], vf, o2[c]);
            }
    }

    // combine, RMSNorm over 64 dims, scale by (1 - LAM0), write f16
    float inv1[8], inv2[8], sq[8], ov[4][8];
#pragma unroll
    for (int r = 0; r < 8; ++r) { inv1[r] = 1.f / l1[r]; inv2[r] = 1.f / l2[r]; sq[r] = 0.f; }
#pragma unroll
    for (int c = 0; c < 4; ++c)
#pragma unroll
        for (int r = 0; r < 8; ++r) {
            float o = o1[c][r] * inv1[r] - lam * (o2[c][r] * inv2[r]);
            ov[c][r] = o; sq[r] += o * o;
        }
#pragma unroll
    for (int r = 0; r < 8; ++r) {
#pragma unroll
        for (int msk = 8; msk >= 1; msk >>= 1) sq[r] += __shfl_xor(sq[r], msk, 32);
        sq[r] = rsqrtf(sq[r] * (1.f / 64.f) + EPS_LN);
    }
#pragma unroll
    for (int c = 0; c < 4; ++c)
#pragma unroll
        for (int r = 0; r < 8; ++r) {
            int col = head * 64 + c * 16 + nl;
            float v = ov[c][r] * sq[r] * subw[c * 16 + nl] * (1.f - LAM0);
            out[(size_t)(row0 + hh * 8 + r) * DMODEL + col] = (_Float16)v;
        }
}

// ------------------------- elementwise / norm kernels -------------------------
__device__ __forceinline__ float blk_sum(float v, float* sbuf) {
#pragma unroll
    for (int m = 16; m >= 1; m >>= 1) v += __shfl_xor(v, m, 32);
    if ((threadIdx.x & 31) == 0) sbuf[threadIdx.x >> 5] = v;
    __syncthreads();
    float t = (threadIdx.x < 8) ? sbuf[threadIdx.x] : 0.f;
#pragma unroll
    for (int m = 4; m >= 1; m >>= 1) t += __shfl_xor(t, m, 32);
    if (threadIdx.x == 0) sbuf[0] = t;
    __syncthreads();
    float r = sbuf[0];
    __syncthreads();
    return r;
}

__global__ void __launch_bounds__(256)
ln_to_half_kernel(const float* __restrict__ x, const float* __restrict__ w,
                  const float* __restrict__ bp, _Float16* __restrict__ out, int D)
{
    __shared__ float sbuf[8];
    const int row = blockIdx.x;
    const float* xr = x + (size_t)row * D;
    float s = 0.f, s2 = 0.f;
    for (int c = threadIdx.x; c < D; c += 256) { float v = xr[c]; s += v; s2 += v * v; }
    s = blk_sum(s, sbuf);  s2 = blk_sum(s2, sbuf);
    float mu = s / D, inv = rsqrtf(s2 / D - mu * mu + EPS_LN);
    _Float16* orow = out + (size_t)row * D;
    for (int c = threadIdx.x; c < D; c += 256)
        orow[c] = (_Float16)((xr[c] - mu) * inv * w[c] + bp[c]);
}

// h = LN(a; n2) + x ; z16 = f16(LN(h; mln1))
__global__ void __launch_bounds__(256)
attn_post_kernel(const float* __restrict__ a, const float* __restrict__ x,
                 const float* __restrict__ n2w, const float* __restrict__ n2b,
                 const float* __restrict__ m1w, const float* __restrict__ m1b,
                 float* __restrict__ h, _Float16* __restrict__ z16)
{
    __shared__ float sbuf[8];
    __shared__ float hbuf[DMODEL];
    const int row = blockIdx.x;
    const float* ar = a + (size_t)row * DMODEL;
    const float* xr = x + (size_t)row * DMODEL;
    float s = 0.f, s2 = 0.f;
    for (int c = threadIdx.x; c < DMODEL; c += 256) { float v = ar[c]; s += v; s2 += v * v; }
    s = blk_sum(s, sbuf);  s2 = blk_sum(s2, sbuf);
    float mu = s / DMODEL, inv = rsqrtf(s2 / DMODEL - mu * mu + EPS_LN);
    float hs = 0.f, hs2 = 0.f;
    for (int c = threadIdx.x; c < DMODEL; c += 256) {
        float hv = (ar[c] - mu) * inv * n2w[c] + n2b[c] + xr[c];
        hbuf[c] = hv; hs += hv; hs2 += hv * hv;
    }
    hs = blk_sum(hs, sbuf);  hs2 = blk_sum(hs2, sbuf);
    float mu2 = hs / DMODEL, inv2 = rsqrtf(hs2 / DMODEL - mu2 * mu2 + EPS_LN);
    float* hr = h + (size_t)row * DMODEL;
    _Float16* zr = z16 + (size_t)row * DMODEL;
    for (int c = threadIdx.x; c < DMODEL; c += 256) {
        float hv = hbuf[c];
        hr[c] = hv;
        zr[c] = (_Float16)((hv - mu2) * inv2 * m1w[c] + m1b[c]);
    }
}

// z2 = f16(LN(gelu(g); mln2)), g already has mb1 added by GEMM epilogue
__global__ void __launch_bounds__(256)
gelu_ln_kernel(const float* __restrict__ g, const float* __restrict__ w,
               const float* __restrict__ bp, _Float16* __restrict__ out)
{
    __shared__ float sbuf[8];
    __shared__ float vbuf[DFF];
    const int row = blockIdx.x;
    const float* gr = g + (size_t)row * DFF;
    float s = 0.f, s2 = 0.f;
    for (int c = threadIdx.x; c < DFF; c += 256) {
        float x = gr[c];
        float v = 0.5f * x * (1.f + erff(x * 0.70710678118f));
        vbuf[c] = v; s += v; s2 += v * v;
    }
    s = blk_sum(s, sbuf);  s2 = blk_sum(s2, sbuf);
    float mu = s / DFF, inv = rsqrtf(s2 / DFF - mu * mu + EPS_LN);
    _Float16* orow = out + (size_t)row * DFF;
    for (int c = threadIdx.x; c < DFF; c += 256)
        orow[c] = (_Float16)((vbuf[c] - mu) * inv * w[c] + bp[c]);
}

__global__ void f32_to_f16_kernel(const float* __restrict__ s, _Float16* __restrict__ d, int n) {
    int i = blockIdx.x * 256 + threadIdx.x;
    if (i < n) d[i] = (_Float16)s[i];
}

// --------------------------------- launcher -----------------------------------
extern "C" void kernel_launch(void* const* d_in, const int* in_sizes, int n_in,
                              void* d_out, int out_size, void* d_ws, size_t ws_size,
                              hipStream_t stream)
{
    const float* x    = (const float*)d_in[0];
    const float* Wq   = (const float*)d_in[1];
    const float* Wk   = (const float*)d_in[2];
    const float* Wv   = (const float*)d_in[3];
    const float* Wo   = (const float*)d_in[4];
    const float* lq1  = (const float*)d_in[5];
    const float* lk1  = (const float*)d_in[6];
    const float* lq2  = (const float*)d_in[7];
    const float* lk2  = (const float*)d_in[8];
    const float* subw = (const float*)d_in[9];
    const float* n1w  = (const float*)d_in[10];
    const float* n1b  = (const float*)d_in[11];
    const float* n2w  = (const float*)d_in[12];
    const float* n2b  = (const float*)d_in[13];
    const float* m1w  = (const float*)d_in[14];
    const float* m1b  = (const float*)d_in[15];
    const float* mW1  = (const float*)d_in[16];
    const float* mb1  = (const float*)d_in[17];
    const float* m2w  = (const float*)d_in[18];
    const float* m2b  = (const float*)d_in[19];
    const float* mW2  = (const float*)d_in[20];
    const float* mb2  = (const float*)d_in[21];
    float* outp = (float*)d_out;

    char* ws = (char*)d_ws;
    size_t off = 0;
    auto carve = [&](size_t bytes) -> char* {
        char* p = ws + off;
        off += (bytes + 255) & ~(size_t)255;
        return p;
    };
    _Float16* wq16  = (_Float16*)carve((size_t)DMODEL * DMODEL * 2);
    _Float16* wk16  = (_Float16*)carve((size_t)DMODEL * DMODEL * 2);
    _Float16* wv16  = (_Float16*)carve((size_t)DMODEL * DMODEL * 2);
    _Float16* wo16  = (_Float16*)carve((size_t)DMODEL * DMODEL * 2);
    _Float16* w1_16 = (_Float16*)carve((size_t)DFF * DMODEL * 2);
    _Float16* w2_16 = (_Float16*)carve((size_t)DMODEL * DFF * 2);
    _Float16* xn16  = (_Float16*)carve((size_t)NROWS * DMODEL * 2);
    _Float16* q16   = (_Float16*)carve((size_t)NROWS * DMODEL * 2);
    _Float16* k16   = (_Float16*)carve((size_t)NROWS * DMODEL * 2);
    _Float16* v16   = (_Float16*)carve((size_t)NROWS * DMODEL * 2);
    _Float16* at16  = (_Float16*)carve((size_t)NROWS * DMODEL * 2);
    float*    a32   = (float*)   carve((size_t)NROWS * DMODEL * 4);
    float*    h32   = (float*)   carve((size_t)NROWS * DMODEL * 4);
    _Float16* z16   = (_Float16*)carve((size_t)NROWS * DMODEL * 2);
    float*    g32   = (float*)   carve((size_t)NROWS * DFF * 4);
    _Float16* z2_16 = (_Float16*)carve((size_t)NROWS * DFF * 2);
    (void)ws_size; (void)in_sizes; (void)n_in; (void)out_size;

    // weights -> f16 (every call; no cached state)
    f32_to_f16_kernel<<<(DMODEL * DMODEL + 255) / 256, 256, 0, stream>>>(Wq, wq16, DMODEL * DMODEL);
    f32_to_f16_kernel<<<(DMODEL * DMODEL + 255) / 256, 256, 0, stream>>>(Wk, wk16, DMODEL * DMODEL);
    f32_to_f16_kernel<<<(DMODEL * DMODEL + 255) / 256, 256, 0, stream>>>(Wv, wv16, DMODEL * DMODEL);
    f32_to_f16_kernel<<<(DMODEL * DMODEL + 255) / 256, 256, 0, stream>>>(Wo, wo16, DMODEL * DMODEL);
    f32_to_f16_kernel<<<(DFF * DMODEL + 255) / 256, 256, 0, stream>>>(mW1, w1_16, DFF * DMODEL);
    f32_to_f16_kernel<<<(DMODEL * DFF + 255) / 256, 256, 0, stream>>>(mW2, w2_16, DMODEL * DFF);

    // xn = f16(LN(x; n1))
    ln_to_half_kernel<<<NROWS, 256, 0, stream>>>(x, n1w, n1b, xn16, DMODEL);

    // QKV projections (f16 out)
    dim3 gD(DMODEL / 128, NROWS / 128);
    gemm_nt_kernel<true, false, false><<<gD, 256, 0, stream>>>(
        xn16, wq16, NROWS, DMODEL, DMODEL, nullptr, q16, nullptr, nullptr);
    gemm_nt_kernel<true, false, false><<<gD, 256, 0, stream>>>(
        xn16, wk16, NROWS, DMODEL, DMODEL, nullptr, k16, nullptr, nullptr);
    gemm_nt_kernel<true, false, false><<<gD, 256, 0, stream>>>(
        xn16, wv16, NROWS, DMODEL, DMODEL, nullptr, v16, nullptr, nullptr);

    // fused differential attention -> at16
    diff_attn_kernel<<<B_SZ * NHEADS * (NSEQ / 128), 256, 0, stream>>>(
        q16, k16, v16, lq1, lk1, lq2, lk2, subw, at16);

    // a = at @ Wo^T (f32)
    gemm_nt_kernel<false, false, false><<<gD, 256, 0, stream>>>(
        at16, wo16, NROWS, DMODEL, DMODEL, a32, nullptr, nullptr, nullptr);

    // h = LN(a; n2) + x ; z = f16(LN(h; mln1))
    attn_post_kernel<<<NROWS, 256, 0, stream>>>(a32, x, n2w, n2b, m1w, m1b, h32, z16);

    // g = z @ mW1^T + mb1 (f32)
    dim3 gF(DFF / 128, NROWS / 128);
    gemm_nt_kernel<false, true, false><<<gF, 256, 0, stream>>>(
        z16, w1_16, NROWS, DFF, DMODEL, g32, nullptr, mb1, nullptr);

    // z2 = f16(LN(gelu(g); mln2))
    gelu_ln_kernel<<<NROWS, 256, 0, stream>>>(g32, m2w, m2b, z2_16);

    // out = z2 @ mW2^T + mb2 + h
    gemm_nt_kernel<false, true, true><<<gD, 256, 0, stream>>>(
        z2_16, w2_16, NROWS, DMODEL, DFF, outp, nullptr, mb2, h32);
}
